// RNN_63316407878385
// MI455X (gfx1250) — matrix operand order repaired and checked by reading, not compiled, not run on hardware
//
#include <hip/hip_runtime.h>
#include <hip/hip_bf16.h>
#include <math.h>

// ---------------------------------------------------------------------------
// Elman RNN for MI455X (gfx1250, wave32, WMMA).
//   L=512, N=128, D=256, H=512
// Kernel 1: xi = x @ w_i + b   (full-chip bf16 WMMA GEMM, writes into d_out)
// Kernel 2: persistent 8-WG kernel, 512 serial steps of
//              h_t = tanh(xi_t + h_{t-1} @ w_h)
//           w_h^T slice LDS-resident for all steps; h kept as a bf16 shadow
//           ping-pong in workspace so per-step LDS staging is an async
//           global->LDS byte copy (ASYNCcnt) instead of f32+convert.
// ---------------------------------------------------------------------------

typedef __attribute__((ext_vector_type(16))) __bf16 v16bf;
typedef __attribute__((ext_vector_type(8)))  __bf16 v8bf;
typedef __attribute__((ext_vector_type(8)))  float  v8f;
typedef __attribute__((ext_vector_type(4)))  int    v4i;

#if defined(__has_builtin)
#  if __has_builtin(__builtin_amdgcn_global_load_async_to_lds_b128)
#    define USE_ASYNC_LDS 1
#  endif
#  if __has_builtin(__builtin_amdgcn_s_wait_asynccnt)
#    define HAVE_WAIT_ASYNC_BUILTIN 1
#  endif
#endif

#ifdef USE_ASYNC_LDS
__device__ __forceinline__ void async_copy16(const void* gsrc, void* ldst) {
    // Param types per hipcc diagnostic: generic pointers to int __vector(4).
    __builtin_amdgcn_global_load_async_to_lds_b128((v4i*)gsrc, (v4i*)ldst, 0, 0);
}
__device__ __forceinline__ void wait_async0() {
#ifdef HAVE_WAIT_ASYNC_BUILTIN
    __builtin_amdgcn_s_wait_asynccnt(0);
#else
    asm volatile("s_wait_asynccnt 0" ::: "memory");
#endif
}
#endif

__device__ __forceinline__ unsigned short f2bf_rne(float f) {
    unsigned int u = __builtin_bit_cast(unsigned int, f);
    unsigned int r = u + 0x7FFFu + ((u >> 16) & 1u);   // round-to-nearest-even
    return (unsigned short)(r >> 16);
}

// A fragment (16x32 bf16, row-major M x K tile in LDS, element stride `stride`)
// lanes 0-15 row M=lane: elems 0..15 = K kbase+0..15
// lanes 16-31 row M=lane-16: K kbase+16..31
__device__ __forceinline__ v16bf load_a_frag(const unsigned short* As, int stride,
                                             int m0, int kbase, int lane) {
    const int m  = m0 + (lane & 15);
    const int kb = kbase + ((lane >> 4) << 4);          // +0 or +16
    const v8bf lo = *reinterpret_cast<const v8bf*>(As + (size_t)m * stride + kb);
    const v8bf hi = *reinterpret_cast<const v8bf*>(As + (size_t)m * stride + kb + 8);
    return __builtin_shufflevector(lo, hi, 0,1,2,3,4,5,6,7,8,9,10,11,12,13,14,15);
}

// B fragment (32x16 bf16). Bs holds W^T: row n (output column), K contiguous.
// lanes 0-15: col n = lane, K = kbase+0..15 ; lanes 16-31: K = kbase+16..31.
__device__ __forceinline__ v16bf load_b_frag(const unsigned short* Bs, int stride,
                                             int n0, int kbase, int lane) {
    const int n  = n0 + (lane & 15);
    const int kb = kbase + ((lane >> 4) << 4);          // +0 or +16
    return *reinterpret_cast<const v16bf*>(Bs + (size_t)n * stride + kb);
}

// ---------------------------------------------------------------------------
// Kernel 1: xi[r, h] = sum_d x[r, d] * w_i[d, h] + b[h],  r = l*N+n (65536 rows)
// Grid: (512, 8) blocks of 256 threads. Per WG: 128-row x 64-col tile, K=256.
// LDS: X tile 128x256 bf16 (64KB) + Wi^T slice 64x256 bf16 (32KB) = 96KB.
// ---------------------------------------------------------------------------
__global__ __launch_bounds__(256) void rnn_xi_gemm(
    const float* __restrict__ X,    // (65536, 256)
    const float* __restrict__ Wi,   // (256, 512)
    const float* __restrict__ bias, // (512)
    float*       __restrict__ XI)   // (65536, 512)  == d_out
{
    constexpr int D = 256, H = 512;
    extern __shared__ unsigned short smem[];
    unsigned short* As = smem;              // 128 x 256
    unsigned short* Bs = smem + 128 * D;    // 64  x 256  (Wi^T slice)

    const int r0  = blockIdx.x * 128;
    const int c0  = blockIdx.y * 64;
    const int tid = threadIdx.x;

    for (int e = tid; e < 128 * D; e += 256) {
        const int row = e >> 8;
        const int d   = e & (D - 1);
        As[e] = f2bf_rne(X[(size_t)(r0 + row) * D + d]);
    }
    for (int e = tid; e < 64 * D; e += 256) {
        const int k = e >> 6;
        const int n = e & 63;
        Bs[n * D + k] = f2bf_rne(Wi[(size_t)k * H + c0 + n]);
    }
    __syncthreads();

    const int wave = tid >> 5;
    const int lane = tid & 31;
    const int m0   = wave * 16;

    v8f c[4] = {};
    for (int kk = 0; kk < D; kk += 32) {
        const v16bf a = load_a_frag(As, D, m0, kk, lane);
#pragma unroll
        for (int nt = 0; nt < 4; ++nt) {
            const v16bf bm = load_b_frag(Bs, D, nt * 16, kk, lane);
            c[nt] = __builtin_amdgcn_wmma_f32_16x16x32_bf16(
                        false, a, false, bm, (short)0, c[nt], false, false);
        }
    }

#pragma unroll
    for (int nt = 0; nt < 4; ++nt) {
        const int   n = c0 + nt * 16 + (lane & 15);
        const float b = bias[n];
#pragma unroll
        for (int vg = 0; vg < 8; ++vg) {
            const int r = r0 + m0 + ((lane >> 4) << 3) + vg;
            XI[(size_t)r * H + n] = c[nt][vg] + b;
        }
    }
}

// ---------------------------------------------------------------------------
// Kernel 2: persistent recurrence. 8 WGs x 256 threads; WG g owns columns
// [64g, 64g+64). Wh^T slice (64x512 bf16 = 64KB) resident in LDS for all
// steps; h_{t-1} arrives via bf16 shadow buffers in d_ws (ping-pong), staged
// into LDS (128KB) with async global->LDS copies. Total LDS 192KB (<320KB).
// ---------------------------------------------------------------------------
__global__ __launch_bounds__(256) void rnn_recurrent(
    const float*  __restrict__ Wh,   // (512, 512)
    float*        __restrict__ OUT,  // (L, N, H): xi on entry, h on exit
    unsigned short* __restrict__ hbf,// bf16 shadow of h: 2 x (N*H) ping-pong
    unsigned int* __restrict__ bar)  // grid barrier counter (zeroed on stream)
{
    constexpr int L = 512, N = 128, H = 512, NB = 8;
    extern __shared__ unsigned short smem[];
    unsigned short* Bs = smem;            // 64  x 512  (Wh^T slice)
    unsigned short* Ah = smem + 64 * H;   // 128 x 512  (h_{t-1} bf16)

    const int tid = threadIdx.x;
    const int c0  = blockIdx.x * 64;

    // Load Wh^T slice once (lives in LDS for all 512 timesteps).
    for (int e = tid; e < 64 * H; e += 256) {
        const int k = e >> 6;
        const int n = e & 63;
        Bs[n * H + k] = f2bf_rne(Wh[(size_t)k * H + c0 + n]);
    }

    // t = 0: h_0 = tanh(xi_0); also publish bf16 shadow into hbf[0].
    for (int e = tid; e < N * 64; e += 256) {
        const int row = e >> 6;
        const int col = c0 + (e & 63);
        const size_t idx = (size_t)row * H + col;
        const float v = tanhf(OUT[idx]);
        OUT[idx] = v;
        hbf[idx] = f2bf_rne(v);
    }
    __threadfence();
    __syncthreads();
    if (tid == 0) {
        atomicAdd(bar, 1u);
        while (__hip_atomic_load(bar, __ATOMIC_RELAXED, __HIP_MEMORY_SCOPE_AGENT) < NB)
            __builtin_amdgcn_s_sleep(1);
    }
    __syncthreads();
    __threadfence();

    const int wave = tid >> 5;
    const int lane = tid & 31;
    const int m0   = wave * 16;

    for (int t = 1; t < L; ++t) {
        // Stage bf16 h_{t-1} shadow -> LDS (pure byte copy, 128KB).
        const unsigned short* hsrc = hbf + (size_t)((t - 1) & 1) * N * H;
#ifdef USE_ASYNC_LDS
        for (int e = tid * 8; e < N * H; e += 256 * 8)      // 16B per lane/op
            async_copy16(hsrc + e, Ah + e);
        wait_async0();
#else
        for (int e = tid * 8; e < N * H; e += 256 * 8) {
            const uint4 v = *reinterpret_cast<const uint4*>(hsrc + e);
            *reinterpret_cast<uint4*>(Ah + e) = v;
        }
#endif
        __syncthreads();

        // Prefetch this lane's 32 xi values so global latency overlaps WMMA.
        float* outt = OUT + (size_t)t * N * H;
        float xiv[4][8];
#pragma unroll
        for (int nt = 0; nt < 4; ++nt) {
            const int n = c0 + nt * 16 + (lane & 15);
#pragma unroll
            for (int vg = 0; vg < 8; ++vg) {
                const int r = m0 + ((lane >> 4) << 3) + vg;
                xiv[nt][vg] = outt[(size_t)r * H + n];
            }
        }

        v8f c[4] = {};
        for (int kk = 0; kk < H; kk += 32) {
            const v16bf a = load_a_frag(Ah, H, m0, kk, lane);
#pragma unroll
            for (int nt = 0; nt < 4; ++nt) {
                const v16bf bm = load_b_frag(Bs, H, nt * 16, kk, lane);
                c[nt] = __builtin_amdgcn_wmma_f32_16x16x32_bf16(
                            false, a, false, bm, (short)0, c[nt], false, false);
            }
        }

        // h_t = tanh(h_{t-1}@Wh + xi_t): write f32 output + bf16 shadow.
        unsigned short* hnext = hbf + (size_t)(t & 1) * N * H;
#pragma unroll
        for (int nt = 0; nt < 4; ++nt) {
            const int n = c0 + nt * 16 + (lane & 15);
#pragma unroll
            for (int vg = 0; vg < 8; ++vg) {
                const int r = m0 + ((lane >> 4) << 3) + vg;
                const size_t idx = (size_t)r * H + n;
                const float v = tanhf(c[nt][vg] + xiv[nt][vg]);
                outt[idx]  = v;
                hnext[idx] = f2bf_rne(v);
            }
        }

        // Grid-wide barrier (monotonic counter, memset to 0 per launch).
        __threadfence();
        __syncthreads();
        if (tid == 0) {
            atomicAdd(bar, 1u);
            const unsigned int target = (unsigned int)(t + 1) * NB;
            while (__hip_atomic_load(bar, __ATOMIC_RELAXED, __HIP_MEMORY_SCOPE_AGENT) < target)
                __builtin_amdgcn_s_sleep(1);
        }
        __syncthreads();
        __threadfence();
    }
}

// ---------------------------------------------------------------------------
extern "C" void kernel_launch(void* const* d_in, const int* in_sizes, int n_in,
                              void* d_out, int out_size, void* d_ws, size_t ws_size,
                              hipStream_t stream) {
    (void)in_sizes; (void)n_in; (void)out_size; (void)ws_size;
    const float* x  = (const float*)d_in[0];   // (512,128,256)
    const float* wi = (const float*)d_in[1];   // (256,512)
    const float* wh = (const float*)d_in[2];   // (512,512)
    const float* bh = (const float*)d_in[3];   // (512)
    float* out = (float*)d_out;                // (512,128,512)

    constexpr int L = 512, N = 128, D = 256, H = 512;

    // Workspace layout: [0,256) barrier counter; then 2 x N*H bf16 h buffers.
    unsigned int*   bar = (unsigned int*)d_ws;
    unsigned short* hbf = (unsigned short*)((char*)d_ws + 256);

    // Allow >64KB dynamic LDS (CDNA5: up to 320KB per workgroup).
    static_cast<void>(hipFuncSetAttribute((const void*)rnn_xi_gemm,
        hipFuncAttributeMaxDynamicSharedMemorySize, (128 * D + 64 * D) * 2));
    static_cast<void>(hipFuncSetAttribute((const void*)rnn_recurrent,
        hipFuncAttributeMaxDynamicSharedMemorySize, (64 * H + N * H) * 2));

    // Phase 1: xi = x @ w_i + b  -> written straight into d_out.
    dim3 g1((L * N) / 128, H / 64);
    size_t sm1 = (size_t)(128 * D + 64 * D) * sizeof(unsigned short); // 96 KB
    hipLaunchKernelGGL(rnn_xi_gemm, g1, dim3(256), sm1, stream, x, wi, bh, out);

    // Zero the grid-barrier counter (capture-safe stream memset).
    static_cast<void>(hipMemsetAsync(d_ws, 0, 256, stream));

    // Phase 2: persistent serial recurrence over d_out.
    size_t sm2 = (size_t)(64 * H + N * H) * sizeof(unsigned short);   // 192 KB
    hipLaunchKernelGGL(rnn_recurrent, dim3(8), dim3(256), sm2, stream,
                       wh, out, hbf, bar);
}
